// PointnetFeatureExtractor_63548336112333
// MI455X (gfx1250) — matrix operand order repaired and checked by reading, not compiled, not run on hardware
//
#include <hip/hip_runtime.h>

#define T 100
#define TPAD 112
#define KNN 8

typedef __attribute__((ext_vector_type(16))) _Float16 v16h;
typedef __attribute__((ext_vector_type(8)))  _Float16 v8h;
typedef __attribute__((ext_vector_type(8)))  float    v8f;

// A-fragment (16x32 f16) from row-major / transposed-activation storage:
// element (m,k) = src[(row0+m)*ldc + k0+k], K contiguous.
// lanes 0-15: M=lane, K in {0..7, 16..23}; lanes 16-31: M=lane-16, K in {8..15, 24..31}
__device__ __forceinline__ v16h fragA_t(const _Float16* src, int ldc, int row0, int k0) {
  int lane = threadIdx.x & 31;
  const _Float16* p = src + (size_t)(row0 + (lane & 15)) * ldc + k0 + ((lane & 16) ? 8 : 0);
  v8h lo = *(const v8h*)p;
  v8h hi = *(const v8h*)(p + 16);
  v16h f;
#pragma unroll
  for (int i = 0; i < 8; ++i) { f[i] = lo[i]; f[i + 8] = hi[i]; }
  return f;
}

// B-fragment (32x16 f16) from transposed activations: element (k,n) = src[(col0+n)*ldc + k0+k]
// lanes 0-15: N=lane, K in {0..7, 16..23}; lanes 16-31: N=lane-16, K in {8..15, 24..31}
__device__ __forceinline__ v16h fragB_t(const _Float16* src, int ldc, int col0, int k0) {
  int lane = threadIdx.x & 31;
  const _Float16* p = src + (size_t)(col0 + (lane & 15)) * ldc + k0 + ((lane & 16) ? 8 : 0);
  v8h lo = *(const v8h*)p;
  v8h hi = *(const v8h*)(p + 16);
  v16h f;
#pragma unroll
  for (int i = 0; i < 8; ++i) { f[i] = lo[i]; f[i + 8] = hi[i]; }
  return f;
}

__device__ __forceinline__ v8f wmma_f16(v16h a, v16h b, v8f c) {
  return __builtin_amdgcn_wmma_f32_16x16x32_f16(false, a, false, b, (short)0, c,
                                                false, false);
}

__device__ __forceinline__ void atomicMaxF(float* addr, float val) {
  int* ia = (int*)addr;
  int old = __float_as_int(*addr);
  while (val > __int_as_float(old)) {
    int assumed = old;
    old = atomicCAS(ia, assumed, __float_as_int(val));
    if (old == assumed) break;
  }
}

// ---------------- prep: fold BN, convert weights to f16 ----------------
__global__ void pn_prep(const float* g1, const float* b1, const float* m1, const float* v1,
                        const float* g2, const float* b2, const float* m2, const float* v2,
                        const float* g3, const float* b3, const float* m3, const float* v3,
                        const float* g4, const float* b4, const float* m4, const float* v4,
                        const float* W2, const float* W3, const float* W4,
                        float* sb, _Float16* w2h, _Float16* w3h, _Float16* w4h) {
  int i = blockIdx.x * blockDim.x + threadIdx.x;
  int stride = gridDim.x * blockDim.x;
  if (i < 32)  { float s = g1[i] * rsqrtf(v1[i] + 1e-5f); sb[i] = s;       sb[32 + i]  = b1[i] - m1[i] * s; }
  if (i < 64)  { float s = g2[i] * rsqrtf(v2[i] + 1e-5f); sb[64 + i] = s;  sb[128 + i] = b2[i] - m2[i] * s; }
  if (i < 96)  { float s = g3[i] * rsqrtf(v3[i] + 1e-5f); sb[192 + i] = s; sb[288 + i] = b3[i] - m3[i] * s; }
  if (i < 128) {
    if (i < 127) { float s = g4[i] * rsqrtf(v4[i] + 1e-5f); sb[384 + i] = s; sb[512 + i] = b4[i] - m4[i] * s; }
    else         { sb[384 + i] = 0.f; sb[512 + i] = 0.f; }
  }
  for (int idx = i; idx < 64 * 32; idx += stride) w2h[idx] = (_Float16)W2[idx];
  for (int idx = i; idx < 8 * 96 * 64; idx += stride) {
    int j = idx / (96 * 64); int r = idx % (96 * 64); int o = r / 64; int c = r % 64;
    w3h[idx] = (_Float16)W3[(o * 64 + c) * 8 + j];          // [j][o][c] <- W3[o][c][j]
  }
  for (int idx = i; idx < 128 * 96; idx += stride) {
    int o = idx / 96; int c = idx % 96;
    w4h[idx] = (o < 127) ? (_Float16)W4[o * 96 + c] : (_Float16)0.f;
  }
}

// ---------------- main: one workgroup per voxel ----------------
__global__ __launch_bounds__(256) void pn_main(
    const float* __restrict__ features, const int* __restrict__ num_voxels,
    const float* __restrict__ W1, const float* __restrict__ sb,
    const _Float16* __restrict__ w2h, const _Float16* __restrict__ w3h,
    const _Float16* __restrict__ w4h, float* __restrict__ out) {
  // LDS map (bytes), lifetime-based reuse. Activations stored TRANSPOSED [n][c]
  // so WMMA fragments and gathers are 16B-contiguous per lane.
  //  [0,1344)      rel  f32[3][112]   (early)     | x3t f16[112][96] (late, [0,21504))
  //  [1344,4928)   idx1 i32[8][112]   (early)
  //  [4928,12096)  x1t  f16[112][32]  (early)
  //  [21504,35840) X2t  f16[112][64]
  //  [35840,36288) sq2  f32[112]
  //  [36288,39872) idx2 i32[8][112]
  //  [39872,54208) gjt  f16[112][64]
  //  [54208,104384) gram f32[112][112]            | ymax f32[128] (late)
  __shared__ __align__(16) char smem[104384];
  __shared__ float meansh[4];
  float*    rel  = (float*)(smem + 0);
  int*      idx1 = (int*)(smem + 1344);
  _Float16* x1t  = (_Float16*)(smem + 4928);
  _Float16* x3t  = (_Float16*)(smem + 0);
  _Float16* X2t  = (_Float16*)(smem + 21504);
  float*    sq2  = (float*)(smem + 35840);
  int*      idx2 = (int*)(smem + 36288);
  _Float16* gjt  = (_Float16*)(smem + 39872);
  float*    gram = (float*)(smem + 54208);
  float*    ymax = (float*)(smem + 54208);

  const int tid  = threadIdx.x;
  const int lane = tid & 31;
  const int wv   = tid >> 5;
  const int vox  = blockIdx.x;
  const float* feat = features + (size_t)vox * T * 4;
  const int nv = num_voxels[vox];
  const float *s1 = sb,       *t1 = sb + 32,  *s2 = sb + 64,  *t2 = sb + 128;
  const float *s3 = sb + 192, *t3 = sb + 288, *s4 = sb + 384, *t4 = sb + 512;

  // ---- mean over all T (faithful: sum all T, divide by nv) ----
  if (tid < 3) {
    float s = 0.f;
    for (int n = 0; n < T; ++n) s += feat[n * 4 + tid];
    meansh[tid] = s / (float)nv;
  }
  __syncthreads();
  for (int f = tid; f < 3 * TPAD; f += 256) {
    int c = f / TPAD, n = f % TPAD;
    rel[f] = (n < T) ? (feat[n * 4 + c] - meansh[c]) : 0.f;
  }
  __syncthreads();

  // ---- knn1: 3-dim, scalar top-8 (nearest-first, ties -> lower idx) ----
  if (tid < T) {
    float bd[KNN]; int bi[KNN];
#pragma unroll
    for (int q = 0; q < KNN; ++q) { bd[q] = 3.0e38f; bi[q] = 0; }
    float xi = rel[tid], yi = rel[TPAD + tid], zi = rel[2 * TPAD + tid];
    for (int j = 0; j < T; ++j) {
      float dx = xi - rel[j], dy = yi - rel[TPAD + j], dz = zi - rel[2 * TPAD + j];
      float d = dx * dx + dy * dy + dz * dz;
      if (d < bd[KNN - 1]) {
        int p = KNN - 1;
        while (p > 0 && d < bd[p - 1]) { bd[p] = bd[p - 1]; bi[p] = bi[p - 1]; --p; }
        bd[p] = d; bi[p] = j;
      }
    }
#pragma unroll
    for (int q = 0; q < KNN; ++q) idx1[q * TPAD + tid] = bi[q];
  }
  __syncthreads();

  // ---- conv1 (K=24, scalar) + relu + BN1 -> x1t[n][o] ----
  for (int f = tid; f < 32 * TPAD; f += 256) {
    int o = f / TPAD, n = f % TPAD;
    float val = 0.f;
    if (n < T) {
      float acc = 0.f;
      for (int j = 0; j < KNN; ++j) {
        int g = idx1[j * TPAD + n];
#pragma unroll
        for (int c = 0; c < 3; ++c) acc += W1[(o * 3 + c) * KNN + j] * rel[c * TPAD + g];
      }
      acc = acc > 0.f ? acc : 0.f;
      val = acc * s1[o] + t1[o];
    }
    x1t[n * 32 + o] = (_Float16)val;
  }
  __syncthreads();

  // ---- conv2: WMMA [64x112] = W2[64x32] * x1[32x112]; relu+BN2 -> X2t[n][m] ----
  for (int t = wv; t < 4 * 7; t += 8) {
    int mt = t / 7, nt = t % 7;
    v8f acc = {};
    acc = wmma_f16(fragA_t(w2h, 32, mt * 16, 0), fragB_t(x1t, 32, nt * 16, 0), acc);
    int n = nt * 16 + (lane & 15);
    int mh = (lane & 16) ? 8 : 0;
    v8h pack;
#pragma unroll
    for (int r = 0; r < 8; ++r) {
      int m = mt * 16 + r + mh;
      float x = acc[r]; x = x > 0.f ? x : 0.f;
      pack[r] = (_Float16)(x * s2[m] + t2[m]);
    }
    *(v8h*)(X2t + n * 64 + mt * 16 + mh) = pack;   // 16B contiguous store
  }
  __syncthreads();

  // ---- sq2[n] = sum_c X2^2 ----
  if (tid < TPAD) {
    float s = 0.f;
    for (int c = 0; c < 64; ++c) { float h = (float)X2t[tid * 64 + c]; s += h * h; }
    sq2[tid] = s;
  }
  __syncthreads();

  // ---- Gram = X2^T X2 via WMMA (49 tiles, K=64); both frags K-contiguous ----
  for (int t = wv; t < 49; t += 8) {
    int it = t / 7, jt = t % 7;
    v8f acc = {};
    for (int k0 = 0; k0 < 64; k0 += 32)
      acc = wmma_f16(fragA_t(X2t, 64, it * 16, k0), fragB_t(X2t, 64, jt * 16, k0), acc);
    int n = jt * 16 + (lane & 15);
    int mh = (lane & 16) ? 8 : 0;
#pragma unroll
    for (int r = 0; r < 8; ++r) gram[(it * 16 + r + mh) * TPAD + n] = acc[r];
  }
  __syncthreads();

  // ---- knn2: dist = sq_i + sq_j - 2*gram, scalar top-8 ----
  if (tid < T) {
    float bd[KNN]; int bi[KNN];
#pragma unroll
    for (int q = 0; q < KNN; ++q) { bd[q] = 3.0e38f; bi[q] = 0; }
    float si = sq2[tid];
    const float* gr = gram + tid * TPAD;
    for (int j = 0; j < T; ++j) {
      float d = si + sq2[j] - 2.f * gr[j];
      if (d < bd[KNN - 1]) {
        int p = KNN - 1;
        while (p > 0 && d < bd[p - 1]) { bd[p] = bd[p - 1]; bi[p] = bi[p - 1]; --p; }
        bd[p] = d; bi[p] = j;
      }
    }
#pragma unroll
    for (int q = 0; q < KNN; ++q) idx2[q * TPAD + tid] = bi[q];
  }
  __syncthreads();
  for (int f = tid; f < 128; f += 256) ymax[f] = -3.0e38f;  // gram dead; reuse as ymax

  // ---- conv3: 42 persistent tiles; per-tap row-gather (b128) then WMMA ----
  v8f zero = {};
  v8f acc3[6];
#pragma unroll
  for (int s = 0; s < 6; ++s) acc3[s] = zero;
  v8h zero8 = {};
  for (int j = 0; j < KNN; ++j) {
    // gjt[n][c] = X2t[idx2[j][n]][c], 128B contiguous per row -> v8h chunks
    for (int f = tid; f < TPAD * 8; f += 256) {
      int n = f >> 3, part = f & 7;
      v8h val = zero8;
      if (n < T) val = *(const v8h*)(X2t + idx2[j * TPAD + n] * 64 + part * 8);
      *(v8h*)(gjt + n * 64 + part * 8) = val;
    }
    __syncthreads();
    const _Float16* w3j = w3h + j * 96 * 64;
    int s = 0;
    for (int t = wv; t < 42; t += 8, ++s) {
      int mt = t / 7, nt = t % 7;
      for (int k0 = 0; k0 < 64; k0 += 32)
        acc3[s] = wmma_f16(fragA_t(w3j, 64, mt * 16, k0),
                           fragB_t(gjt, 64, nt * 16, k0), acc3[s]);
    }
    __syncthreads();
  }
  {
    int s = 0;
    for (int t = wv; t < 42; t += 8, ++s) {
      int mt = t / 7, nt = t % 7;
      int n = nt * 16 + (lane & 15);
      int mh = (lane & 16) ? 8 : 0;
      v8h pack;
#pragma unroll
      for (int r = 0; r < 8; ++r) {
        int m = mt * 16 + r + mh;
        float x = acc3[s][r]; x = x > 0.f ? x : 0.f;
        x = x * s3[m] + t3[m];
        pack[r] = (n < T) ? (_Float16)x : (_Float16)0.f;
      }
      *(v8h*)(x3t + n * 96 + mt * 16 + mh) = pack;  // 16B contiguous store
    }
  }
  __syncthreads();

  // ---- conv4: WMMA [128x112] = W4[128x96] * x3[96x112]; relu+BN4; masked max ----
  for (int t = wv; t < 56; t += 8) {
    int mt = t / 7, nt = t % 7;
    v8f acc = {};
    for (int k0 = 0; k0 < 96; k0 += 32)
      acc = wmma_f16(fragA_t(w4h, 96, mt * 16, k0), fragB_t(x3t, 96, nt * 16, k0), acc);
    int n = nt * 16 + (lane & 15);
    int mh = (lane & 16) ? 8 : 0;
    if (n < T) {
#pragma unroll
      for (int r = 0; r < 8; ++r) {
        int m = mt * 16 + r + mh;
        if (m < 127) {
          float x = acc[r]; x = x > 0.f ? x : 0.f;
          x = x * s4[m] + t4[m];
          atomicMaxF(&ymax[m], (n < nv) ? x : 0.f);   // mask-then-max, faithful
        }
      }
    }
  }
  __syncthreads();

  if (tid < 127) out[(size_t)vox * 128 + tid] = ymax[tid];
  if (tid == 127) {
    float mi = -3.0e38f;
    for (int n = 0; n < T; ++n) { float x = feat[n * 4 + 3]; mi = mi > x ? mi : x; }
    out[(size_t)vox * 128 + 127] = mi;
  }
}

extern "C" void kernel_launch(void* const* d_in, const int* in_sizes, int n_in,
                              void* d_out, int out_size, void* d_ws, size_t ws_size,
                              hipStream_t stream) {
  const float* features   = (const float*)d_in[0];
  const int*   num_voxels = (const int*)d_in[1];
  const float* W1 = (const float*)d_in[2];
  const float* g1 = (const float*)d_in[3];
  const float* b1 = (const float*)d_in[4];
  const float* m1 = (const float*)d_in[5];
  const float* v1 = (const float*)d_in[6];
  const float* W2 = (const float*)d_in[7];
  const float* g2 = (const float*)d_in[8];
  const float* b2 = (const float*)d_in[9];
  const float* m2 = (const float*)d_in[10];
  const float* v2 = (const float*)d_in[11];
  const float* W3 = (const float*)d_in[12];
  const float* g3 = (const float*)d_in[13];
  const float* b3 = (const float*)d_in[14];
  const float* m3 = (const float*)d_in[15];
  const float* v3 = (const float*)d_in[16];
  const float* W4 = (const float*)d_in[17];
  const float* g4 = (const float*)d_in[18];
  const float* b4 = (const float*)d_in[19];
  const float* m4 = (const float*)d_in[20];
  const float* v4 = (const float*)d_in[21];

  char* ws = (char*)d_ws;
  float*    sbuf = (float*)ws;                    // 640 f32  = 2560 B
  _Float16* w2h  = (_Float16*)(ws + 2560);        // 64*32    = 4096 B
  _Float16* w3h  = (_Float16*)(ws + 6656);        // 8*96*64  = 98304 B
  _Float16* w4h  = (_Float16*)(ws + 104960);      // 128*96   = 24576 B

  pn_prep<<<64, 256, 0, stream>>>(g1, b1, m1, v1, g2, b2, m2, v2,
                                  g3, b3, m3, v3, g4, b4, m4, v4,
                                  W2, W3, W4, sbuf, w2h, w3h, w4h);
  pn_main<<<6000, 256, 0, stream>>>(features, num_voxels, W1, sbuf,
                                    w2h, w3h, w4h, (float*)d_out);
}
